// sphred_17437567222075
// MI455X (gfx1250) — matrix operand order repaired and checked by reading, not compiled, not run on hardware
//
#include <hip/hip_runtime.h>
#include <hip/hip_bf16.h>

// ---------------------------------------------------------------------------
// Model dims
// ---------------------------------------------------------------------------
#define Hn   300        // hidden
#define G3   900        // 3*H
#define Bn   64         // batch
#define Tn   40         // seq len (TU == TT)
#define Vn   19495      // vocab
#define Mfull (Tn*Bn)   // 2560

// padded dims for WMMA (bf16, K tiles of 32, N tiles of 64)
#define KP   320        // K pad for K=300
#define KZ   608        // K pad for K=600
#define NG   960        // N pad for N=900
#define NZ   320        // N pad for N=300
#define NV   19520      // N pad for N=19495

// GEMM block tiling
#define MW   32         // rows per wave
#define BM   128        // rows per block (4 waves stacked on M)
#define BN   64         // cols per block (shared B tile via LDS)

typedef __attribute__((ext_vector_type(16))) __bf16 v16bf;
typedef __attribute__((ext_vector_type(8)))  __bf16 v8bf;
typedef __attribute__((ext_vector_type(8)))  float  v8f;

__device__ __forceinline__ float sigm(float x) { return 1.0f / (1.0f + __expf(-x)); }

// ---------------------------------------------------------------------------
// WMMA GEMM: C[M,N] = A[M,Kpad](bf16,row-major) * W[Npad,Kpad](bf16)^T (+bias)
//
// Block = 4 waves stacked on M (128x64 tile).  The 64-col x 32-K slice of W
// that all 4 waves share is double-buffered through LDS (8 KB), cutting B
// read traffic 4x.  Each wave computes 32x64 = 8 accumulator tiles with
// 8 v_wmma_f32_16x16x32_bf16 per K-step.  All 4 B fragments are pulled into
// distinct registers up front so the 8 ds_loads pipeline ahead of the WMMA
// chain; A fragments come straight from global (rows are private per wave).
// ---------------------------------------------------------------------------
__global__ __launch_bounds__(128)
void wmma_gemm_bf16(const __bf16* __restrict__ A, int lda,
                    const __bf16* __restrict__ W, int ldw,
                    float* __restrict__ C, long long ldc,
                    const float* __restrict__ bias,
                    int M, int N, int Kpad)
{
    __shared__ __align__(32) __bf16 Bs[2][BN * 32];

    const int tid    = threadIdx.x;
    const int lane   = tid & 31;
    const int wv     = tid >> 5;
    const int laneLo = lane & 15;
    const int laneHi = lane >> 4;

    const int nBase = blockIdx.x * BN;
    int mBase = blockIdx.y * BM + wv * MW;
    const bool live = (mBase < M);          // small GEMMs: dead waves still
    if (!live) mBase = 0;                   // compute (EXEC all-1s) + barrier

    // LDS staging assignment: 128 threads move 64 rows x 32 halfs (4 KB)
    const int srow  = tid >> 1;
    const int shalf = (tid & 1) * 16;
    const __bf16* wsrc = W + (size_t)(nBase + srow) * ldw + shalf;
    __bf16* sdst = &Bs[0][srow * 32 + shalf];

    v8f acc0[4] = {};
    v8f acc1[4] = {};
    const __bf16* Ar0 = A + (size_t)(mBase + laneLo)      * lda;
    const __bf16* Ar1 = A + (size_t)(mBase + 16 + laneLo) * lda;

    // prologue: stage k=0 slice into buffer 0
    *(v16bf*)sdst = *(const v16bf*)wsrc;
    __syncthreads();

    int buf = 0;
    for (int k = 0; k < Kpad; k += 32, buf ^= 1) {
        // A fragments first (global, long latency).  contiguous k order used here:
        // lanes 0-15 hold K[0:16), lanes 16-31 hold K[16:32) (matches B).
        v8bf a0l = *(const v8bf*)(Ar0 + k +     laneHi * 16);
        v8bf a0h = *(const v8bf*)(Ar0 + k + 8 + laneHi * 16);
        v8bf a1l = *(const v8bf*)(Ar1 + k +     laneHi * 16);
        v8bf a1h = *(const v8bf*)(Ar1 + k + 8 + laneHi * 16);

        // prefetch next K-slice into the other LDS buffer
        if (k + 32 < Kpad)
            *(v16bf*)(sdst + (buf ^ 1) * (BN * 32)) = *(const v16bf*)(wsrc + k + 32);

        // all 4 B fragments into distinct registers so the ds_loads pipeline
        const __bf16* bsrc = &Bs[buf][laneLo * 32 + laneHi * 16];
        v16bf bf[4];
        #pragma unroll
        for (int t = 0; t < 4; ++t)
            bf[t] = *(const v16bf*)(bsrc + t * 16 * 32);

        v16bf a0 = __builtin_shufflevector(a0l, a0h, 0,1,2,3,4,5,6,7,8,9,10,11,12,13,14,15);
        v16bf a1 = __builtin_shufflevector(a1l, a1h, 0,1,2,3,4,5,6,7,8,9,10,11,12,13,14,15);

        #pragma unroll
        for (int t = 0; t < 4; ++t) {
            acc0[t] = __builtin_amdgcn_wmma_f32_16x16x32_bf16(
                          false, a0, false, bf[t], (short)0, acc0[t], false, false);
            acc1[t] = __builtin_amdgcn_wmma_f32_16x16x32_bf16(
                          false, a1, false, bf[t], (short)0, acc1[t], false, false);
        }
        __syncthreads();   // compute(buf) + prefetch(buf^1) done before reuse
    }

    if (!live) return;
    // C layout: VGPR r -> row mBase + laneHi*8 + r, col nBase + laneLo
    #pragma unroll
    for (int t = 0; t < 4; ++t) {
        int n = nBase + t * 16 + laneLo;
        if (n < N) {
            float bv = bias ? bias[n] : 0.0f;
            int r0 = mBase + laneHi * 8;
            #pragma unroll
            for (int r = 0; r < 8; ++r) {
                C[(long long)(r0 + r)      * ldc + n] = acc0[t][r] + bv;
                C[(long long)(r0 + 16 + r) * ldc + n] = acc1[t][r] + bv;
            }
        }
    }
}

// ---------------------------------------------------------------------------
// f32 [N,K] weight -> zero-padded bf16 [Npad,Kpad]
// ---------------------------------------------------------------------------
__global__ void cvt_weight(const float* __restrict__ src, __bf16* __restrict__ dst,
                           int N, int K, int Npad, int Kpad)
{
    int idx = blockIdx.x * blockDim.x + threadIdx.x;
    if (idx >= Npad * Kpad) return;
    int n = idx / Kpad, k = idx % Kpad;
    dst[idx] = (n < N && k < K) ? (__bf16)src[n * K + k] : (__bf16)0.0f;
}

// emb gather -> bf16 A matrix, row m = t*B + b, padded K
__global__ void gather_emb(const float* __restrict__ emb, const int* __restrict__ ids,
                           __bf16* __restrict__ dst)
{
    int idx = blockIdx.x * blockDim.x + threadIdx.x;
    if (idx >= Mfull * KP) return;
    int m = idx / KP, k = idx % KP;
    int t = m / Bn, b = m % Bn;
    __bf16 v = (__bf16)0.0f;
    if (k < Hn) {
        int id = ids[b * Tn + t];
        v = (__bf16)emb[(size_t)id * Hn + k];
    }
    dst[idx] = v;
}

// f32 [Bn,Hn] -> bf16 [Bn,KP] padded
__global__ void cvt_act(const float* __restrict__ src, __bf16* __restrict__ dst)
{
    int idx = blockIdx.x * blockDim.x + threadIdx.x;
    if (idx >= Bn * KP) return;
    int b = idx / KP, k = idx % KP;
    dst[idx] = (k < Hn) ? (__bf16)src[b * Hn + k] : (__bf16)0.0f;
}

__global__ void zero_h(float* __restrict__ h, __bf16* __restrict__ hbf)
{
    int idx = blockIdx.x * blockDim.x + threadIdx.x;
    if (idx >= Bn * KP) return;
    int b = idx / KP, k = idx % KP;
    hbf[idx] = (__bf16)0.0f;
    if (k < Hn) h[b * Hn + k] = 0.0f;
}

// ---------------------------------------------------------------------------
// Fused GRU gate step (torch r,z,n order); optional final-state capture at
// t==lens[b]-1 and bf16 hidden-state emission for the logits GEMM.
// ---------------------------------------------------------------------------
__global__ void gru_gate(const float* __restrict__ gi, const float* __restrict__ gh,
                         const float* __restrict__ bih, const float* __restrict__ bhh,
                         float* __restrict__ h, __bf16* __restrict__ hbf,
                         int t, const int* __restrict__ lens,
                         float* __restrict__ hfinal, __bf16* __restrict__ hs_bf)
{
    int idx = blockIdx.x * blockDim.x + threadIdx.x;
    if (idx >= Bn * Hn) return;
    int b = idx / Hn, j = idx % Hn;
    const float* gir = gi + (size_t)(t * Bn + b) * NG;
    const float* ghr = gh + (size_t)b * NG;

    float ir = gir[j]            + bih[j];
    float iz = gir[Hn + j]       + bih[Hn + j];
    float in = gir[2 * Hn + j]   + bih[2 * Hn + j];
    float hr = ghr[j]            + bhh[j];
    float hz = ghr[Hn + j]       + bhh[Hn + j];
    float hn = ghr[2 * Hn + j]   + bhh[2 * Hn + j];

    float r = sigm(ir + hr);
    float z = sigm(iz + hz);
    float n = tanhf(in + r * hn);
    float hold = h[idx];
    float hnew = (1.0f - z) * n + z * hold;

    h[idx] = hnew;
    hbf[b * KP + j] = (__bf16)hnew;
    if (lens   && t == lens[b] - 1) hfinal[idx] = hnew;
    if (hs_bf) hs_bf[(size_t)(t * Bn + b) * KP + j] = (__bf16)hnew;
}

// Session GRU cell with h0 = 0: gh reduces to bhh, h' = (1-z)*n.
__global__ void sess_gate(const float* __restrict__ gi,
                          const float* __restrict__ bih, const float* __restrict__ bhh,
                          float* __restrict__ out)
{
    int idx = blockIdx.x * blockDim.x + threadIdx.x;
    if (idx >= Bn * Hn) return;
    int b = idx / Hn, j = idx % Hn;
    const float* gir = gi + (size_t)b * NG;
    float ir = gir[j]          + bih[j];
    float iz = gir[Hn + j]     + bih[Hn + j];
    float in = gir[2 * Hn + j] + bih[2 * Hn + j];
    float r = sigm(ir + bhh[j]);
    float z = sigm(iz + bhh[Hn + j]);
    float n = tanhf(in + r * bhh[2 * Hn + j]);
    out[idx] = (1.0f - z) * n;
}

// z_hidden = interleave(sA, sB) -> bf16 [Bn, KZ] padded
__global__ void interleave_z(const float* __restrict__ sA, const float* __restrict__ sB,
                             __bf16* __restrict__ zbf)
{
    int idx = blockIdx.x * blockDim.x + threadIdx.x;
    if (idx >= Bn * KZ) return;
    int b = idx / KZ, j = idx % KZ;
    __bf16 v = (__bf16)0.0f;
    if (j < 2 * Hn) {
        const float* s = (j & 1) ? sB : sA;
        v = (__bf16)s[b * Hn + (j >> 1)];
    }
    zbf[idx] = v;
}

// dec_h0 = z_mean + exp(z_log_var/2) * eps  (reads from d_out region)
__global__ void reparam(const float* __restrict__ zm, const float* __restrict__ zlv,
                        const float* __restrict__ eps,
                        float* __restrict__ h, __bf16* __restrict__ hbf)
{
    int idx = blockIdx.x * blockDim.x + threadIdx.x;
    if (idx >= Bn * KP) return;
    int b = idx / KP, k = idx % KP;
    if (k < Hn) {
        float v = zm[b * Hn + k] + __expf(0.5f * zlv[b * Hn + k]) * eps[b * Hn + k];
        h[b * Hn + k] = v;
        hbf[idx] = (__bf16)v;
    } else {
        hbf[idx] = (__bf16)0.0f;
    }
}

// ---------------------------------------------------------------------------
// Host-side orchestration
// ---------------------------------------------------------------------------
static inline void gemm(hipStream_t s, const __bf16* A, int lda, const __bf16* W, int ldw,
                        float* C, long long ldc, const float* bias,
                        int M, int N, int Npad, int Kpad)
{
    dim3 grid(Npad / BN, (M + BM - 1) / BM);
    wmma_gemm_bf16<<<grid, 128, 0, s>>>(A, lda, W, ldw, C, ldc, bias, M, N, Kpad);
}

static inline int eb(int n) { return (n + 255) / 256; }   // elementwise blocks

extern "C" void kernel_launch(void* const* d_in, const int* in_sizes, int n_in,
                              void* d_out, int out_size, void* d_ws, size_t ws_size,
                              hipStream_t stream)
{
    (void)in_sizes; (void)n_in; (void)out_size; (void)ws_size;

    // ---- inputs (setup_inputs order, tuples flattened) ----
    const float* emb  = (const float*)d_in[0];
    const float* Wih[5], *Whh[5], *bih[5], *bhh[5];       // encA,encB,sessA,sessB,dec
    for (int i = 0; i < 5; ++i) {
        Wih[i] = (const float*)d_in[1 + 4 * i + 0];
        Whh[i] = (const float*)d_in[1 + 4 * i + 1];
        bih[i] = (const float*)d_in[1 + 4 * i + 2];
        bhh[i] = (const float*)d_in[1 + 4 * i + 3];
    }
    const float* Wm   = (const float*)d_in[21];
    const float* bm   = (const float*)d_in[22];
    const float* Wv   = (const float*)d_in[23];
    const float* bv   = (const float*)d_in[24];
    const float* Wout = (const float*)d_in[25];
    const float* bout = (const float*)d_in[26];
    const float* eps  = (const float*)d_in[27];
    const int* u1_ids = (const int*)d_in[28];
    const int* u1_len = (const int*)d_in[29];
    const int* u2_ids = (const int*)d_in[30];
    const int* u2_len = (const int*)d_in[31];
    const int* tg_ids = (const int*)d_in[32];

    float* out = (float*)d_out;
    const long long ZM_OFF  = (long long)Tn * Bn * Vn;
    const long long ZLV_OFF = ZM_OFF + (long long)Bn * Hn;

    // ---- workspace bump allocator ----
    size_t off = 0;
    auto alloc = [&](size_t bytes) -> void* {
        void* p = (char*)d_ws + off;
        off += (bytes + 255) & ~(size_t)255;
        return p;
    };
    __bf16* w_ih[5], *w_hh[5];
    for (int i = 0; i < 5; ++i) {
        w_ih[i] = (__bf16*)alloc((size_t)NG * KP * 2);
        w_hh[i] = (__bf16*)alloc((size_t)NG * KP * 2);
    }
    __bf16* w_m   = (__bf16*)alloc((size_t)NZ * KZ * 2);
    __bf16* w_v   = (__bf16*)alloc((size_t)NZ * KZ * 2);
    __bf16* w_out = (__bf16*)alloc((size_t)NV * KP * 2);
    __bf16* a_emb = (__bf16*)alloc((size_t)Mfull * KP * 2);
    __bf16* hs_bf = (__bf16*)alloc((size_t)Mfull * KP * 2);
    __bf16* h_bf  = (__bf16*)alloc((size_t)Bn * KP * 2);
    __bf16* uh_bf = (__bf16*)alloc((size_t)Bn * KP * 2);
    __bf16* z_bf  = (__bf16*)alloc((size_t)Bn * KZ * 2);
    float* gi   = (float*)alloc((size_t)Mfull * NG * 4);
    float* gh   = (float*)alloc((size_t)Bn * NG * 4);     // also reused as session gi
    float* h    = (float*)alloc((size_t)Bn * Hn * 4);
    float* u1_h = (float*)alloc((size_t)Bn * Hn * 4);
    float* u2_h = (float*)alloc((size_t)Bn * Hn * 4);
    float* sA   = (float*)alloc((size_t)Bn * Hn * 4);
    float* sB   = (float*)alloc((size_t)Bn * Hn * 4);

    // ---- 1. convert all weights to padded bf16 (once per call) ----
    for (int i = 0; i < 5; ++i) {
        cvt_weight<<<eb(NG * KP), 256, 0, stream>>>(Wih[i], w_ih[i], G3, Hn, NG, KP);
        cvt_weight<<<eb(NG * KP), 256, 0, stream>>>(Whh[i], w_hh[i], G3, Hn, NG, KP);
    }
    cvt_weight<<<eb(NZ * KZ), 256, 0, stream>>>(Wm,   w_m,   Hn, 2 * Hn, NZ, KZ);
    cvt_weight<<<eb(NZ * KZ), 256, 0, stream>>>(Wv,   w_v,   Hn, 2 * Hn, NZ, KZ);
    cvt_weight<<<eb(NV * KP), 256, 0, stream>>>(Wout, w_out, Vn, Hn,     NV, KP);

    // ---- 2. the two utterance encoders ----
    const int* enc_ids[2]  = { u1_ids, u2_ids };
    const int* enc_lens[2] = { u1_len, u2_len };
    float*     enc_out[2]  = { u1_h, u2_h };
    for (int e = 0; e < 2; ++e) {
        gather_emb<<<eb(Mfull * KP), 256, 0, stream>>>(emb, enc_ids[e], a_emb);
        gemm(stream, a_emb, KP, w_ih[e], KP, gi, NG, nullptr, Mfull, G3, NG, KP);
        zero_h<<<eb(Bn * KP), 256, 0, stream>>>(h, h_bf);
        for (int t = 0; t < Tn; ++t) {
            gemm(stream, h_bf, KP, w_hh[e], KP, gh, NG, nullptr, Bn, G3, NG, KP);
            gru_gate<<<eb(Bn * Hn), 256, 0, stream>>>(gi, gh, bih[e], bhh[e], h, h_bf,
                                                      t, enc_lens[e], enc_out[e], nullptr);
        }
    }

    // ---- 3. session GRU cells (h0 = 0) ----
    cvt_act<<<eb(Bn * KP), 256, 0, stream>>>(u1_h, uh_bf);
    gemm(stream, uh_bf, KP, w_ih[2], KP, gh, NG, nullptr, Bn, G3, NG, KP);
    sess_gate<<<eb(Bn * Hn), 256, 0, stream>>>(gh, bih[2], bhh[2], sA);
    cvt_act<<<eb(Bn * KP), 256, 0, stream>>>(u2_h, uh_bf);
    gemm(stream, uh_bf, KP, w_ih[3], KP, gh, NG, nullptr, Bn, G3, NG, KP);
    sess_gate<<<eb(Bn * Hn), 256, 0, stream>>>(gh, bih[3], bhh[3], sB);

    // ---- 4. latent: z_mean / z_log_var straight into d_out, then reparam ----
    interleave_z<<<eb(Bn * KZ), 256, 0, stream>>>(sA, sB, z_bf);
    gemm(stream, z_bf, KZ, w_m, KZ, out + ZM_OFF,  Hn, bm, Bn, Hn, NZ, KZ);
    gemm(stream, z_bf, KZ, w_v, KZ, out + ZLV_OFF, Hn, bv, Bn, Hn, NZ, KZ);
    reparam<<<eb(Bn * KP), 256, 0, stream>>>(out + ZM_OFF, out + ZLV_OFF, eps, h, h_bf);

    // ---- 5. decoder GRU over target, collecting bf16 hidden states ----
    gather_emb<<<eb(Mfull * KP), 256, 0, stream>>>(emb, tg_ids, a_emb);
    gemm(stream, a_emb, KP, w_ih[4], KP, gi, NG, nullptr, Mfull, G3, NG, KP);
    hipMemsetAsync(hs_bf, 0, (size_t)Mfull * KP * 2, stream);
    for (int t = 0; t < Tn; ++t) {
        gemm(stream, h_bf, KP, w_hh[4], KP, gh, NG, nullptr, Bn, G3, NG, KP);
        gru_gate<<<eb(Bn * Hn), 256, 0, stream>>>(gi, gh, bih[4], bhh[4], h, h_bf,
                                                  t, nullptr, nullptr, hs_bf);
    }

    // ---- 6. the big one: logits = hs @ Wout^T + bout ----
    gemm(stream, hs_bf, KP, w_out, KP, out, Vn, bout, Mfull, Vn, NV, KP);
}
